// BahdanauEncoderDecoderLSTM_76553497084058
// MI455X (gfx1250) — compile-verified, matrix operand order checked by reading, not run on hardware
//
#include <hip/hip_runtime.h>
#include <math.h>

#define F_  128
#define HE_ 512
#define HD_ 512
#define A_  256
#define B_  32
#define T_  512

typedef __attribute__((ext_vector_type(16))) __bf16 v16bf;
typedef __attribute__((ext_vector_type(8)))  float  v8f;

// ---------------- workspace layout (byte offsets, all 256-aligned) ----------------
#define OFF_SYNC    0                          // int[64] barrier counters
#define OFF_XPACK   256                        // bf16 [T][2][4][512]   x in A-frag order
#define OFF_WENCP   (OFF_XPACK  + 4194304)     // bf16 [20][128][512]   [Wih;Whh]^T B-frag
#define OFF_WDECP   (OFF_WENCP  + 2621440)     // bf16 [32][128][512]   folded dec W B-frag
#define OFF_WEAP    (OFF_WDECP  + 4194304)     // bf16 [16][16][512]    Wenc_w^T B-frag
#define OFF_WDAP    (OFF_WEAP   + 262144)      // bf16 [16][16][512]    Wdec_w^T B-frag
#define OFF_EBIAS   (OFF_WDAP   + 262144)      // f32 [2048]
#define OFF_DBIAS   (OFF_EBIAS  + 8192)        // f32 [2048]
#define OFF_HPACKE  (OFF_DBIAS  + 8192)        // bf16 [2][16][512]     enc h A-frag
#define OFF_CENC    (OFF_HPACKE + 32768)       // f32 [32][512]
#define OFF_ZENC    (OFF_CENC   + 65536)       // f32 [32][2048]
#define OFF_ENCSBF  (OFF_ZENC   + 262144)      // bf16 [32][512][512]   enc states plain
#define OFF_ENCPACK (OFF_ENCSBF + 16777216)    // bf16 [1024][16][512]  enc states A-frag
#define OFF_EPROJ   (OFF_ENCPACK+ 16777216)    // f32 [16384][256]      enc_proj + Wenc_b
#define OFF_HPACKD  (OFF_EPROJ  + 16777216)    // bf16 [2][16][512]     dec h A-frag
#define OFF_CDEC    (OFF_HPACKD + 32768)       // f32 [32][512]
#define OFF_ZDEC    (OFF_CDEC   + 65536)       // f32 [32][2048]
#define OFF_DSC     (OFF_ZDEC   + 262144)      // f32 [32][256]  dscore + Wdec_b + attn_bias
#define OFF_LOGITS  (OFF_DSC    + 32768)       // f32 [32][512]
#define OFF_WSM     (OFF_LOGITS + 65536)       // f32 [32][512]  softmax weights
#define OFF_CTXP    (OFF_WSM    + 65536)       // bf16 [2][16][512]  ctx A-frag

// ---------------- helpers ----------------
__device__ __forceinline__ unsigned short f2bf(float f) {
  unsigned u = __float_as_uint(f);
  unsigned r = u + 0x7FFFu + ((u >> 16) & 1u);   // round-to-nearest-even
  return (unsigned short)(r >> 16);
}
__device__ __forceinline__ float bf2f(unsigned short h) {
  return __uint_as_float(((unsigned)h) << 16);
}
__device__ __forceinline__ float sigf(float x) { return 1.0f / (1.0f + __expf(-x)); }

// fragment element k' (0..31) <-> (e, half):  k' = 16*(e/8) + (e%8) + 8*half
__device__ __forceinline__ int frag_e(int kk)    { return ((kk >> 4) << 3) | (kk & 7); }
__device__ __forceinline__ int frag_half(int kk) { return (kk >> 3) & 1; }

// CDNA5 async global->LDS copy (ASYNCcnt); 16B per lane, INST_OFFSET applies to both sides
__device__ __forceinline__ void async_b128(unsigned lds_addr, const void* gptr) {
  asm volatile("global_load_async_to_lds_b128 %0, %1, off"
               :: "v"(lds_addr), "v"((unsigned long long)(size_t)gptr) : "memory");
}
__device__ __forceinline__ void async_b128_off16(unsigned lds_addr, const void* gptr) {
  asm volatile("global_load_async_to_lds_b128 %0, %1, off offset:16"
               :: "v"(lds_addr), "v"((unsigned long long)(size_t)gptr) : "memory");
}
__device__ __forceinline__ void wait_async0() {
  asm volatile("s_wait_asynccnt 0" ::: "memory");
}

// device-wide barrier for a persistent kernel (all NWG blocks resident)
__device__ __forceinline__ void grid_sync(int* cnt, int* gen, int nwg, int target) {
  __threadfence();
  __syncthreads();
  if (threadIdx.x == 0) {
    if (atomicAdd(cnt, 1) == nwg - 1) {
      atomicExch(cnt, 0);
      __threadfence();
      atomicAdd(gen, 1);
    }
    while (__hip_atomic_load(gen, __ATOMIC_ACQUIRE, __HIP_MEMORY_SCOPE_AGENT) < target) {
      __builtin_amdgcn_s_sleep(1);
    }
  }
  __syncthreads();
}

__device__ __forceinline__ v8f wmma_bf16(v16bf a, v16bf b, v8f c) {
  return __builtin_amdgcn_wmma_f32_16x16x32_bf16(false, a, false, b, (short)0, c,
                                                 false, false);
}

// ---------------- prep: zero state, combine biases, decoder h0/c0 ----------------
__global__ void k_prep(char* ws, const float* ebih, const float* ebhh,
                       const float* dbih, const float* dbhh) {
  int tid = blockIdx.x * 256 + threadIdx.x;          // 0..16383
  int* syncp = (int*)(ws + OFF_SYNC);
  unsigned short* hpackE = (unsigned short*)(ws + OFF_HPACKE);
  unsigned short* hpackD = (unsigned short*)(ws + OFF_HPACKD);
  float* cenc  = (float*)(ws + OFF_CENC);
  float* cdec  = (float*)(ws + OFF_CDEC);
  float* ebias = (float*)(ws + OFF_EBIAS);
  float* dbias = (float*)(ws + OFF_DBIAS);
  if (tid < 64) syncp[tid] = 0;
  if (tid < 2048) {
    ebias[tid] = ebih[tid] + ebhh[tid];
    dbias[tid] = dbih[tid] + dbhh[tid];
  }
  hpackE[tid] = 0;      // h_enc = 0
  cenc[tid]   = 0.0f;
  // decoder initial step from zero input/state: c0 = sig(bi)*tanh(bg), h0 = sig(bo)*tanh(c0)
  int b = tid >> 9, j = tid & 511;
  float bi = dbih[j]        + dbhh[j];
  float bg = dbih[1024 + j] + dbhh[1024 + j];
  float bo = dbih[1536 + j] + dbhh[1536 + j];
  float c0 = sigf(bi) * tanhf(bg);
  float h0 = sigf(bo) * tanhf(c0);
  cdec[tid] = c0;
  int kt = j >> 5, kk = j & 31;
  hpackD[(((b >> 4) * 16 + kt) << 9) + (((b & 15) + (frag_half(kk) << 4)) << 4) + frag_e(kk)]
      = f2bf(h0);
}

// ---------------- pack x into A-fragment order ----------------
__global__ void k_pack_x(char* ws, const float* x) {
  int idx = blockIdx.x * 256 + threadIdx.x;          // 0 .. 2^21-1
  unsigned short* xpack = (unsigned short*)(ws + OFF_XPACK);
  int e  = idx & 15;
  int L  = (idx >> 4) & 31;
  int kt = (idx >> 9) & 3;
  int mt = (idx >> 11) & 1;
  int t  = idx >> 12;
  int m  = mt * 16 + (L & 15);                       // batch row
  int k  = kt * 32 + (((e >> 3) << 4) | (e & 7) | ((L >> 4) << 3));
  xpack[idx] = f2bf(x[(m * T_ + t) * F_ + k]);
}

// ---------------- pack encoder weight [Wih;Whh]^T (K=640) into B-fragment order ----
__global__ void k_pack_wenc(char* ws, const float* wih, const float* whh) {
  int idx = blockIdx.x * 256 + threadIdx.x;          // 0 .. 20*128*512-1
  unsigned short* wencp = (unsigned short*)(ws + OFF_WENCP);
  int e  = idx & 15;
  int L  = (idx >> 4) & 31;
  int nt = (idx >> 9) & 127;
  int kt = idx >> 16;                                // 0..19
  int n  = nt * 16 + (L & 15);
  int k  = kt * 32 + (((e >> 3) << 4) | (e & 7) | ((L >> 4) << 3));
  float v = (k < F_) ? wih[n * F_ + k] : whh[n * HE_ + (k - F_)];
  wencp[idx] = f2bf(v);
}

// ---------- pack decoder weight: dec_Wih with dec_Whh folded into cols 512..1023 ----
__global__ void k_pack_wdec(char* ws, const float* dwih, const float* dwhh) {
  int idx = blockIdx.x * 256 + threadIdx.x;          // 0 .. 32*128*512-1
  unsigned short* wdecp = (unsigned short*)(ws + OFF_WDECP);
  int e  = idx & 15;
  int L  = (idx >> 4) & 31;
  int nt = (idx >> 9) & 127;
  int kt = idx >> 16;                                // 0..31
  int n  = nt * 16 + (L & 15);
  int k  = kt * 32 + (((e >> 3) << 4) | (e & 7) | ((L >> 4) << 3));
  float v = dwih[n * 1024 + k];
  if (k >= 512) v += dwhh[n * HD_ + (k - 512)];
  wdecp[idx] = f2bf(v);
}

// ---------------- pack attention projections Wenc_w^T / Wdec_w^T ----------------
__global__ void k_pack_attn(char* ws, const float* wencw, const float* wdecw) {
  int idx = blockIdx.x * 256 + threadIdx.x;          // 0 .. 2*131072-1
  int sel  = idx >> 17;
  int idx2 = idx & 131071;
  unsigned short* dst = (unsigned short*)(ws + (sel ? OFF_WDAP : OFF_WEAP));
  const float* src = sel ? wdecw : wencw;
  int e  = idx2 & 15;
  int L  = (idx2 >> 4) & 31;
  int nt = (idx2 >> 9) & 15;
  int kt = idx2 >> 13;                               // 0..15
  int a  = nt * 16 + (L & 15);
  int k  = kt * 32 + (((e >> 3) << 4) | (e & 7) | ((L >> 4) << 3));
  dst[idx2] = f2bf(src[a * HE_ + k]);
}

// ---------------- persistent encoder: 32 WGs x 8 waves = 256 waves = 256 tiles ----
__global__ void __launch_bounds__(256, 1) k_encoder(char* ws) {
  // B tiles kt=0..15 pinned in LDS per wave (16KB/wave, 128KB/WG); kt=16..19 from L2
  __shared__ unsigned short bsh[8 * 16 * 512];
  const int lane  = threadIdx.x & 31;
  const int wave  = threadIdx.x >> 5;
  const int gwave = blockIdx.x * 8 + wave;           // 0..255
  int* syncp = (int*)(ws + OFF_SYNC);
  const unsigned short* xpack  = (const unsigned short*)(ws + OFF_XPACK);
  const unsigned short* wencp  = (const unsigned short*)(ws + OFF_WENCP);
  unsigned short* hpackE       = (unsigned short*)(ws + OFF_HPACKE);
  unsigned short* encsbf       = (unsigned short*)(ws + OFF_ENCSBF);
  unsigned short* encpack      = (unsigned short*)(ws + OFF_ENCPACK);
  float* zenc = (float*)(ws + OFF_ZENC);
  float* cenc = (float*)(ws + OFF_CENC);
  const float* ebias = (const float*)(ws + OFF_EBIAS);
  const int mt = gwave >> 7, nt = gwave & 127;
  unsigned short* myb = bsh + (wave << 13);          // 16*512 elems per wave
  // stage invariant B fragments into LDS via async copies (ASYNCcnt)
  for (int kt = 0; kt < 16; ++kt) {
    const unsigned short* g = wencp + ((((kt << 7) + nt) << 9) + (lane << 4));
    unsigned la = (unsigned)(size_t)(myb + (kt << 9) + (lane << 4));
    async_b128(la, g);
    async_b128_off16(la, g);
  }
  wait_async0();
  int tgt = 0;
  for (int t = 0; t < T_; ++t) {
    // z = [x_t | h] @ W  (M=32, N=2048, K=640)
    v8f acc = {0.f, 0.f, 0.f, 0.f, 0.f, 0.f, 0.f, 0.f};
    for (int kt = 0; kt < 20; ++kt) {
      const unsigned short* ap = (kt < 4)
          ? xpack  + (((t * 2 + mt) * 4 + kt) << 9)
          : hpackE + ((mt * 16 + (kt - 4)) << 9);
      v16bf a = *(const v16bf*)(ap + (lane << 4));
      v16bf b = (kt < 16)
          ? *(const v16bf*)(myb + (kt << 9) + (lane << 4))
          : *(const v16bf*)(wencp + ((((kt << 7) + nt) << 9) + (lane << 4)));
      acc = wmma_bf16(a, b, acc);
    }
    {
      int col = lane & 15, half = lane >> 4;
#pragma unroll
      for (int v = 0; v < 8; ++v)
        zenc[(mt * 16 + v + 8 * half) * 2048 + (nt << 4) + col] = acc[v];
    }
    grid_sync(&syncp[0], &syncp[1], 32, ++tgt);
    // gates + state update + write h in plain & fragment layouts
    for (int u = 0; u < 2; ++u) {
      int idx = (blockIdx.x << 8) + threadIdx.x + (u << 13);   // 0..16383
      int b = idx >> 9, j = idx & 511;
      const float* z = zenc + b * 2048;
      float i_ = sigf (z[j]          + ebias[j]);
      float f_ = sigf (z[512 + j]    + ebias[512 + j]);
      float g_ = tanhf(z[1024 + j]   + ebias[1024 + j]);
      float o_ = sigf (z[1536 + j]   + ebias[1536 + j]);
      float c  = f_ * cenc[idx] + i_ * g_;
      cenc[idx] = c;
      float h = o_ * tanhf(c);
      unsigned short hb = f2bf(h);
      encsbf[(((b << 9) + t) << 9) + j] = hb;
      int kt = j >> 5, kk = j & 31;
      int e = frag_e(kk), hf = frag_half(kk);
      hpackE[(((b >> 4) * 16 + kt) << 9) + (((b & 15) + (hf << 4)) << 4) + e] = hb;
      int row = (b << 9) + t;                        // b*T + t
      encpack[(((row >> 4) * 16 + kt) << 9) + (((row & 15) + (hf << 4)) << 4) + e] = hb;
    }
    grid_sync(&syncp[0], &syncp[1], 32, ++tgt);
  }
}

// ---------------- enc_proj = enc_states @ Wenc_w^T + Wenc_b  (16384x512x256) ------
__global__ void __launch_bounds__(256) k_encproj(char* ws, const float* wenc_b) {
  const int lane  = threadIdx.x & 31;
  const int wave  = threadIdx.x >> 5;
  const int gwave = blockIdx.x * 8 + wave;           // 0..16383 tiles
  const unsigned short* encpack = (const unsigned short*)(ws + OFF_ENCPACK);
  const unsigned short* weap    = (const unsigned short*)(ws + OFF_WEAP);
  float* eproj = (float*)(ws + OFF_EPROJ);
  int mt = gwave >> 4, nt = gwave & 15;
  v8f acc = {0.f, 0.f, 0.f, 0.f, 0.f, 0.f, 0.f, 0.f};
  for (int kt = 0; kt < 16; ++kt) {
    v16bf a = *(const v16bf*)(encpack + (((mt * 16 + kt) << 9) + (lane << 4)));
    v16bf b = *(const v16bf*)(weap    + ((((kt << 4) + nt) << 9) + (lane << 4)));
    acc = wmma_bf16(a, b, acc);
  }
  int col = lane & 15, half = lane >> 4;
#pragma unroll
  for (int v = 0; v < 8; ++v) {
    int row = mt * 16 + v + 8 * half;
    int a_  = (nt << 4) + col;
    eproj[row * 256 + a_] = acc[v] + wenc_b[a_];
  }
}

// ---------------- persistent decoder: 64 WGs x 8 waves -----------------------------
__global__ void __launch_bounds__(256, 1) k_decoder(char* ws, float* out,
                                                    const float* wdec_b,
                                                    const float* attn_bias,
                                                    const float* v_w,
                                                    const float* v_b) {
  // S5 B tiles kt=0..15 pinned in LDS per wave (128KB/WG); kt=16..31 from L2
  __shared__ unsigned short bshD[8 * 16 * 512];
  __shared__ float sred[256];
  const int lane  = threadIdx.x & 31;
  const int wave  = threadIdx.x >> 5;
  const int gwave = blockIdx.x * 8 + wave;           // 0..511
  int* syncp = (int*)(ws + OFF_SYNC);
  int* cnt = &syncp[2]; int* gen = &syncp[3];
  unsigned short* hpackD = (unsigned short*)(ws + OFF_HPACKD);
  unsigned short* ctxp   = (unsigned short*)(ws + OFF_CTXP);
  const unsigned short* wdap   = (const unsigned short*)(ws + OFF_WDAP);
  const unsigned short* wdecp  = (const unsigned short*)(ws + OFF_WDECP);
  const unsigned short* encsbf = (const unsigned short*)(ws + OFF_ENCSBF);
  const float* eproj = (const float*)(ws + OFF_EPROJ);
  const float* dbias = (const float*)(ws + OFF_DBIAS);
  float* cdec   = (float*)(ws + OFF_CDEC);
  float* zdec   = (float*)(ws + OFF_ZDEC);
  float* dsc    = (float*)(ws + OFF_DSC);
  float* logits = (float*)(ws + OFF_LOGITS);
  float* wsm    = (float*)(ws + OFF_WSM);
  // stage S5 weight fragments (only meaningful for gwave<256; others stage harmlessly)
  {
    int nt5 = gwave & 127, mt5ok = (gwave < 256);
    unsigned short* myb = bshD + (wave << 13);
    for (int kt = 0; kt < 16; ++kt) {
      const unsigned short* g =
          wdecp + ((((kt << 7) + (mt5ok ? nt5 : 0)) << 9) + (lane << 4));
      unsigned la = (unsigned)(size_t)(myb + (kt << 9) + (lane << 4));
      async_b128(la, g);
      async_b128_off16(la, g);
    }
    wait_async0();
  }
  int tgt = 0;
  for (int t = 0; t < T_; ++t) {
    // S1: dscore = h @ Wdec_w^T + Wdec_b + attn_bias   (32 tiles)
    if (gwave < 32) {
      int mt = gwave >> 4, nt = gwave & 15;
      v8f acc = {0.f, 0.f, 0.f, 0.f, 0.f, 0.f, 0.f, 0.f};
      for (int kt = 0; kt < 16; ++kt) {
        v16bf a = *(const v16bf*)(hpackD + (((mt * 16 + kt) << 9) + (lane << 4)));
        v16bf b = *(const v16bf*)(wdap   + ((((kt << 4) + nt) << 9) + (lane << 4)));
        acc = wmma_bf16(a, b, acc);
      }
      int col = lane & 15, half = lane >> 4;
#pragma unroll
      for (int v = 0; v < 8; ++v) {
        int m  = mt * 16 + v + 8 * half;
        int a_ = (nt << 4) + col;
        dsc[m * 256 + a_] = acc[v] + wdec_b[a_] + attn_bias[a_];
      }
    }
    grid_sync(cnt, gen, 64, ++tgt);
    // S2: logits[b][tt] = sum_a tanh(eproj + dsc) * Vw[a] + Vb   (one wave per pair)
    for (int p = 0; p < 32; ++p) {
      int pair = (gwave << 5) + p;                   // 0..16383
      int b = pair >> 9, tt = pair & 511;
      const float* ep = eproj + (((b << 9) + tt) << 8);
      const float* ds = dsc + (b << 8);
      float sum = 0.f;
      for (int a0 = lane; a0 < 256; a0 += 32)
        sum += tanhf(ep[a0] + ds[a0]) * v_w[a0];
#pragma unroll
      for (int off = 16; off; off >>= 1) sum += __shfl_xor(sum, off, 32);
      if (lane == 0) logits[(b << 9) + tt] = sum + v_b[0];
    }
    grid_sync(cnt, gen, 64, ++tgt);
    // S3: softmax over tt (one WG per batch row)
    if (blockIdx.x < 32) {
      int b = blockIdx.x;
      float* lg = logits + (b << 9);
      float* w  = wsm + (b << 9);
      float m = -3.4e38f;
      for (int i = threadIdx.x; i < 512; i += 256) m = fmaxf(m, lg[i]);
      sred[threadIdx.x] = m; __syncthreads();
      for (int s = 128; s; s >>= 1) {
        if (threadIdx.x < s) sred[threadIdx.x] = fmaxf(sred[threadIdx.x], sred[threadIdx.x + s]);
        __syncthreads();
      }
      float mx = sred[0]; __syncthreads();
      float sum = 0.f;
      for (int i = threadIdx.x; i < 512; i += 256) {
        float e0 = __expf(lg[i] - mx);
        w[i] = e0; sum += e0;
      }
      sred[threadIdx.x] = sum; __syncthreads();
      for (int s = 128; s; s >>= 1) {
        if (threadIdx.x < s) sred[threadIdx.x] += sred[threadIdx.x + s];
        __syncthreads();
      }
      float inv = 1.0f / sred[0]; __syncthreads();
      for (int i = threadIdx.x; i < 512; i += 256) w[i] *= inv;
    }
    grid_sync(cnt, gen, 64, ++tgt);
    // S4: ctx[b][e] = sum_tt w[b][tt]*enc[b][tt][e]  -> packed A-fragments
    {
      int idx = (blockIdx.x << 8) + threadIdx.x;     // 0..16383
      int b = idx >> 9, e0 = idx & 511;
      const float* w = wsm + (b << 9);
      const unsigned short* hs = encsbf + (((size_t)(b << 9)) << 9) + e0;
      float acc0 = 0.f;
      for (int tt = 0; tt < 512; ++tt) acc0 += w[tt] * bf2f(hs[tt << 9]);
      int kt = e0 >> 5, kk = e0 & 31;
      ctxp[(((b >> 4) * 16 + kt) << 9) + (((b & 15) + (frag_half(kk) << 4)) << 4) + frag_e(kk)]
          = f2bf(acc0);
    }
    grid_sync(cnt, gen, 64, ++tgt);
    // S5: z = [ctx | h] @ W''  (M=32, N=2048, K=1024; 256 tiles)
    if (gwave < 256) {
      int mt = gwave >> 7, nt = gwave & 127;
      unsigned short* myb = bshD + (wave << 13);
      v8f acc = {0.f, 0.f, 0.f, 0.f, 0.f, 0.f, 0.f, 0.f};
      for (int kt = 0; kt < 32; ++kt) {
        const unsigned short* ap = (kt < 16)
            ? ctxp   + ((mt * 16 + kt) << 9)
            : hpackD + ((mt * 16 + (kt - 16)) << 9);
        v16bf a = *(const v16bf*)(ap + (lane << 4));
        v16bf b = (kt < 16)
            ? *(const v16bf*)(myb + (kt << 9) + (lane << 4))
            : *(const v16bf*)(wdecp + ((((kt << 7) + nt) << 9) + (lane << 4)));
        acc = wmma_bf16(a, b, acc);
      }
      int col = lane & 15, half = lane >> 4;
#pragma unroll
      for (int v = 0; v < 8; ++v)
        zdec[(mt * 16 + v + 8 * half) * 2048 + (nt << 4) + col] = acc[v];
    }
    grid_sync(cnt, gen, 64, ++tgt);
    // S6: gates, state update, output h, repack h for next step
    {
      int idx = (blockIdx.x << 8) + threadIdx.x;     // 0..16383
      int b = idx >> 9, j = idx & 511;
      const float* z = zdec + b * 2048;
      float i_ = sigf (z[j]        + dbias[j]);
      float f_ = sigf (z[512 + j]  + dbias[512 + j]);
      float g_ = tanhf(z[1024 + j] + dbias[1024 + j]);
      float o_ = sigf (z[1536 + j] + dbias[1536 + j]);
      float c  = f_ * cdec[idx] + i_ * g_;
      cdec[idx] = c;
      float h = o_ * tanhf(c);
      out[(((size_t)(b << 9) + t) << 9) + j] = h;
      int kt = j >> 5, kk = j & 31;
      hpackD[(((b >> 4) * 16 + kt) << 9) + (((b & 15) + (frag_half(kk) << 4)) << 4) + frag_e(kk)]
          = f2bf(h);
    }
    grid_sync(cnt, gen, 64, ++tgt);
  }
}

extern "C" void kernel_launch(void* const* d_in, const int* in_sizes, int n_in,
                              void* d_out, int out_size, void* d_ws, size_t ws_size,
                              hipStream_t stream) {
  char* ws = (char*)d_ws;
  const float* x        = (const float*)d_in[0];
  const float* enc_Wih  = (const float*)d_in[1];
  const float* enc_Whh  = (const float*)d_in[2];
  const float* enc_bih  = (const float*)d_in[3];
  const float* enc_bhh  = (const float*)d_in[4];
  const float* Wenc_w   = (const float*)d_in[5];
  const float* Wenc_b   = (const float*)d_in[6];
  const float* Wdec_w   = (const float*)d_in[7];
  const float* Wdec_b   = (const float*)d_in[8];
  const float* V_w      = (const float*)d_in[9];
  const float* V_b      = (const float*)d_in[10];
  const float* attn_b   = (const float*)d_in[11];
  const float* dec_Wih  = (const float*)d_in[12];
  const float* dec_Whh  = (const float*)d_in[13];
  const float* dec_bih  = (const float*)d_in[14];
  const float* dec_bhh  = (const float*)d_in[15];

  k_prep     <<<64,   256, 0, stream>>>(ws, enc_bih, enc_bhh, dec_bih, dec_bhh);
  k_pack_x   <<<8192, 256, 0, stream>>>(ws, x);
  k_pack_wenc<<<5120, 256, 0, stream>>>(ws, enc_Wih, enc_Whh);
  k_pack_wdec<<<8192, 256, 0, stream>>>(ws, dec_Wih, dec_Whh);
  k_pack_attn<<<1024, 256, 0, stream>>>(ws, Wenc_w, Wdec_w);
  k_encoder  <<<32,   256, 0, stream>>>(ws);
  k_encproj  <<<2048, 256, 0, stream>>>(ws, Wenc_b);
  k_decoder  <<<64,   256, 0, stream>>>(ws, (float*)d_out, Wdec_b, attn_b, V_w, V_b);
}
